// RNN_42348377538861
// MI455X (gfx1250) — matrix operand order repaired and checked by reading, not compiled, not run on hardware
//
#include <hip/hip_runtime.h>
#include <hip/hip_bf16.h>

// ---------------------------------------------------------------------------
// RNN: xw = x@W_ih^T + b_ih (one big GEMM)
//      h_t = tanh(xw_t + h_{t-1}@W_hh^T + b_hh)   (512 sequential step GEMMs)
//      o   = h_all@W_dec^T + b_dec               (one big batched GEMM)
// GEMMs run on v_wmma_f32_16x16x32_bf16; the big GEMM stages its shared
// B tile in LDS (double-buffered, async global->LDS) and software-pipelines
// its A fragments in registers. tanh uses gfx1250 v_tanh_f32 when available.
// ---------------------------------------------------------------------------

typedef __bf16 v16bf __attribute__((ext_vector_type(16)));
typedef __bf16 v8bf  __attribute__((ext_vector_type(8)));
typedef float  v8f   __attribute__((ext_vector_type(8)));
typedef int    v4i   __attribute__((ext_vector_type(4)));

union BF16x16 { v16bf v; v8bf h[2]; };

static constexpr int Bsz = 64;
static constexpr int T   = 512;
static constexpr int I   = 1024;
static constexpr int H   = 2048;
static constexpr int O   = 1024;

#define LDS_AS __attribute__((address_space(3)))
#define GLB_AS __attribute__((address_space(1)))

#if defined(__has_builtin)
#  if __has_builtin(__builtin_amdgcn_global_load_async_to_lds_b128)
#    define USE_ASYNC_LDS 1
#  endif
#endif
#ifndef USE_ASYNC_LDS
#  define USE_ASYNC_LDS 0
#endif

#if defined(__has_builtin) && __has_builtin(__builtin_amdgcn_s_wait_asynccnt)
#  define WAIT_ASYNC(n) __builtin_amdgcn_s_wait_asynccnt(n)
#else
#  define WAIT_ASYNC(n) asm volatile("s_wait_asynccnt %0" :: "n"(n) : "memory")
#endif

#if defined(__has_builtin) && __has_builtin(__builtin_amdgcn_tanhf)
#  define FAST_TANH(x) __builtin_amdgcn_tanhf(x)
#elif defined(__has_builtin) && __has_builtin(__builtin_amdgcn_tanh_f32)
#  define FAST_TANH(x) __builtin_amdgcn_tanh_f32(x)
#else
#  define FAST_TANH(x) tanhf(x)
#endif

// ---------------------------------------------------------------------------
// fp32 -> bf16 conversion (vectorized by 4)
// ---------------------------------------------------------------------------
__global__ __launch_bounds__(256)
void f32_to_bf16_kernel(const float* __restrict__ src,
                        __hip_bfloat16* __restrict__ dst, int n) {
  int i = (blockIdx.x * blockDim.x + threadIdx.x) * 4;
  if (i + 3 < n) {
    float4 f = *(const float4*)(src + i);
    dst[i + 0] = __float2bfloat16(f.x);
    dst[i + 1] = __float2bfloat16(f.y);
    dst[i + 2] = __float2bfloat16(f.z);
    dst[i + 3] = __float2bfloat16(f.w);
  } else {
    for (int j = i; j < n; ++j) dst[j] = __float2bfloat16(src[j]);
  }
}

// ---------------------------------------------------------------------------
// Generic bf16 GEMM:  C[M,N] = A[M,K] * W[N,K]^T + bias[N]
// Workgroup = 8 waves; wave w -> rows [blockIdx.y*128 + 16w, +16),
// columns [blockIdx.x*64, +64) as 4 accumulator tiles of 16x16.
// B tile (64 W-rows x 64 K) staged in LDS, double-buffered with async
// global->LDS loads; A fragments double-buffered in registers so global
// latency overlaps the WMMA block.
// Fragment layouts (CDNA5, wave32):
//   A 16x32 bf16: lane l -> row (l&15); halves 0..7 = K[kb..kb+8),
//                 halves 8..15 = K[kb+16..kb+24), kb = 8*(l>>4)
//   B 32x16 bf16: lane l -> col (l&15); halves in the same K order as A
//   C/D f32:      elem r -> (m = r + 8*(l>>4), n = l&15)
// ---------------------------------------------------------------------------
static constexpr int KB   = 64;  // K per LDS stage (two WMMA k-steps)
static constexpr int LROW = 80;  // padded halfs per LDS row (160 B)

__global__ __launch_bounds__(256)
void gemm_bf16_kernel(const __hip_bfloat16* __restrict__ A_, long lda,
                      const __hip_bfloat16* __restrict__ W_, long ldb,
                      const float* __restrict__ bias,
                      float* __restrict__ C, long ldc, int K) {
  __shared__ __bf16 lB[2][64 * LROW];  // 2 x 10 KB

  const __bf16* A = (const __bf16*)A_;
  const __bf16* W = (const __bf16*)W_;

  const int tid  = threadIdx.x;
  const int lane = tid & 31;
  const int wave = tid >> 5;
  const int ml   = lane & 15;
  const int hi   = lane >> 4;

  const int mrow_blk = blockIdx.y * 128 + wave * 16;
  const int ncol_blk = blockIdx.x * 64;

  const __bf16* arow = A + (long)(mrow_blk + ml) * lda + (hi * 8);

  // Cooperative staging: thread -> (row = tid/4, 32B chunk = tid%4) of tile.
  const int srow = tid >> 2;
  const int scol = (tid & 3) * 16;
  const __bf16* wsrc = W + (long)(ncol_blk + srow) * ldb + scol;
  __bf16* ld0 = &lB[0][srow * LROW + scol];
  __bf16* ld1 = &lB[1][srow * LROW + scol];

  auto stage = [&](int buf, int kb) {
    __bf16* d = buf ? ld1 : ld0;
#if USE_ASYNC_LDS
    __builtin_amdgcn_global_load_async_to_lds_b128(
        (GLB_AS v4i*)(wsrc + kb), (LDS_AS v4i*)d, 0, 0);
    __builtin_amdgcn_global_load_async_to_lds_b128(
        (GLB_AS v4i*)(wsrc + kb + 8), (LDS_AS v4i*)(d + 8), 0, 0);
#else
    v8bf t0 = *(const v8bf*)(wsrc + kb);
    v8bf t1 = *(const v8bf*)(wsrc + kb + 8);
    *(v8bf*)d = t0;
    *(v8bf*)(d + 8) = t1;
#endif
  };

  auto loadA = [&](int kb, BF16x16& x0, BF16x16& x1) {
    x0.h[0] = *(const v8bf*)(arow + kb);
    x0.h[1] = *(const v8bf*)(arow + kb + 16);
    x1.h[0] = *(const v8bf*)(arow + kb + 32);
    x1.h[1] = *(const v8bf*)(arow + kb + 48);
  };

  v8f acc[4] = {};

  stage(0, 0);
  BF16x16 a0, a1;
  loadA(0, a0, a1);

  int buf = 0;
  for (int kb = 0; kb < K; kb += KB) {
    const bool havenext = (kb + KB < K);
    if (havenext) {
      stage(buf ^ 1, kb + KB);
#if USE_ASYNC_LDS
      WAIT_ASYNC(2);  // 2 newest = next tile; current tile complete
#endif
    } else {
#if USE_ASYNC_LDS
      WAIT_ASYNC(0);
#endif
    }
    __syncthreads();

    // Issue next A-block loads; they overlap the 8 WMMAs below.
    BF16x16 n0, n1;
    if (havenext) {
      __builtin_prefetch(arow + kb + KB + 256, 0, 0);  // global_prefetch_b8
      loadA(kb + KB, n0, n1);
    }

    const __bf16* lbase = &lB[buf][ml * LROW + hi * 8];
#pragma unroll
    for (int ks = 0; ks < KB; ks += 32) {
      const BF16x16& a = (ks == 0) ? a0 : a1;
      BF16x16 b[4];
#pragma unroll
      for (int nt = 0; nt < 4; ++nt) {
        const __bf16* lrow = lbase + nt * 16 * LROW + ks;
        b[nt].h[0] = *(const v8bf*)(lrow);
        b[nt].h[1] = *(const v8bf*)(lrow + 16);
      }
#pragma unroll
      for (int nt = 0; nt < 4; ++nt)
        acc[nt] = __builtin_amdgcn_wmma_f32_16x16x32_bf16(
            false, a.v, false, b[nt].v, (short)0, acc[nt], false, false);
    }
    __syncthreads();

    if (havenext) { a0 = n0; a1 = n1; }
    buf ^= 1;
  }

#pragma unroll
  for (int nt = 0; nt < 4; ++nt) {
    const int n = ncol_blk + nt * 16 + ml;
    const float bv = bias[n];
#pragma unroll
    for (int r = 0; r < 8; ++r) {
      const int m = mrow_blk + r + 8 * hi;
      C[(long)m * ldc + n] = acc[nt][r] + bv;
    }
  }
}

// ---------------------------------------------------------------------------
// One recurrence step:
//   h_t[b,n] = tanh( xw[b,t,n] + sum_k h_{t-1}[b,k]*W_hh[n,k] + b_hh[n] )
// M=64 (batch), N=2048. 8 waves = 4 M-tiles x 2 N-halves; each wave 16x64.
// Latency-bound; W_hh (8 MB bf16) stays hot in L2, so plain global loads.
// ---------------------------------------------------------------------------
__global__ __launch_bounds__(256)
void rnn_step_kernel(const __hip_bfloat16* __restrict__ hprev_, long lda,
                     const __hip_bfloat16* __restrict__ Whh_,
                     const float* __restrict__ bhh,
                     const float* __restrict__ xw_t,
                     __hip_bfloat16* __restrict__ hall_t,
                     float* __restrict__ hfinal) {
  const __bf16* A = (const __bf16*)hprev_;
  const __bf16* W = (const __bf16*)Whh_;

  const int lane = threadIdx.x & 31;
  const int wave = threadIdx.x >> 5;
  const int ml   = lane & 15;
  const int hi   = lane >> 4;

  const int m_tile = wave & 3;   // 4 tiles x 16 rows = all 64 batch rows
  const int n_half = wave >> 2;  // 2 halves of 64 columns
  const int ncol_blk = blockIdx.x * 128 + n_half * 64;

  const __bf16* arow = A + (long)(m_tile * 16 + ml) * lda + (hi * 8);

  const __bf16* wrow[4];
#pragma unroll
  for (int nt = 0; nt < 4; ++nt)
    wrow[nt] = W + (long)(ncol_blk + nt * 16 + ml) * (long)H + (hi * 8);

  v8f acc[4] = {};

  for (int k = 0; k < H; k += 32) {
    __builtin_prefetch(arow + k + 256, 0, 0);

    BF16x16 a;
    a.h[0] = *(const v8bf*)(arow + k);
    a.h[1] = *(const v8bf*)(arow + k + 16);

    BF16x16 b[4];
#pragma unroll
    for (int nt = 0; nt < 4; ++nt) {
      b[nt].h[0] = *(const v8bf*)(wrow[nt] + k);
      b[nt].h[1] = *(const v8bf*)(wrow[nt] + k + 16);
    }
#pragma unroll
    for (int nt = 0; nt < 4; ++nt)
      acc[nt] = __builtin_amdgcn_wmma_f32_16x16x32_bf16(
          false, a.v, false, b[nt].v, (short)0, acc[nt], false, false);
  }

#pragma unroll
  for (int nt = 0; nt < 4; ++nt) {
    const int n = ncol_blk + nt * 16 + ml;
    const float bv = bhh[n];
#pragma unroll
    for (int r = 0; r < 8; ++r) {
      const int b = m_tile * 16 + r + 8 * hi;
      const float z = acc[nt][r] + xw_t[(long)b * T * H + n] + bv;
      const float hv = FAST_TANH(z);
      hall_t[(long)b * T * H + n] = __float2bfloat16(hv);
      if (hfinal) hfinal[(long)b * H + n] = hv;
    }
  }
}

// ---------------------------------------------------------------------------
// Host-side orchestration
// ---------------------------------------------------------------------------
extern "C" void kernel_launch(void* const* d_in, const int* in_sizes, int n_in,
                              void* d_out, int out_size, void* d_ws, size_t ws_size,
                              hipStream_t stream) {
  const float* x     = (const float*)d_in[0];
  const float* h0    = (const float*)d_in[1];
  const float* W_ih  = (const float*)d_in[2];
  const float* b_ih  = (const float*)d_in[3];
  const float* W_hh  = (const float*)d_in[4];
  const float* b_hh  = (const float*)d_in[5];
  const float* W_dec = (const float*)d_in[6];
  const float* b_dec = (const float*)d_in[7];

  float* out_o = (float*)d_out;                    // [B,T,O]
  float* out_h = out_o + (size_t)Bsz * T * O;      // [B,H]

  // Workspace carve-up
  char* ws = (char*)d_ws;
  __hip_bfloat16* x_bf   = (__hip_bfloat16*)ws;  ws += (size_t)Bsz * T * I * 2;
  __hip_bfloat16* wih_bf = (__hip_bfloat16*)ws;  ws += (size_t)H * I * 2;
  __hip_bfloat16* whh_bf = (__hip_bfloat16*)ws;  ws += (size_t)H * H * 2;
  __hip_bfloat16* wdec_bf= (__hip_bfloat16*)ws;  ws += (size_t)O * H * 2;
  __hip_bfloat16* h0_bf  = (__hip_bfloat16*)ws;  ws += (size_t)Bsz * H * 2;
  float*          xw     = (float*)ws;           ws += (size_t)Bsz * T * H * 4;
  __hip_bfloat16* hall   = (__hip_bfloat16*)ws;  // [B,T,H] bf16

  auto conv = [&](const float* s, __hip_bfloat16* d, size_t n) {
    int blocks = (int)((n + 1023) / 1024);
    f32_to_bf16_kernel<<<blocks, 256, 0, stream>>>(s, d, (int)n);
  };
  conv(x,     x_bf,   (size_t)Bsz * T * I);
  conv(W_ih,  wih_bf, (size_t)H * I);
  conv(W_hh,  whh_bf, (size_t)H * H);
  conv(W_dec, wdec_bf,(size_t)O * H);
  conv(h0,    h0_bf,  (size_t)Bsz * H);

  // Phase 1: xw[B*T, H] = x[B*T, I] @ W_ih[H, I]^T + b_ih
  gemm_bf16_kernel<<<dim3(H / 64, (Bsz * T) / 128), 256, 0, stream>>>(
      x_bf, (long)I, wih_bf, (long)I, b_ih, xw, (long)H, I);

  // Phase 2: sequential recurrence over T steps
  for (int t = 0; t < T; ++t) {
    const __hip_bfloat16* hprev = t ? (hall + (size_t)(t - 1) * H) : h0_bf;
    long lda = t ? (long)T * H : (long)H;
    rnn_step_kernel<<<dim3(H / 128), 256, 0, stream>>>(
        hprev, lda, whh_bf, b_hh,
        xw + (size_t)t * H, hall + (size_t)t * H,
        (t == T - 1) ? out_h : nullptr);
  }

  // Phase 3: o[B*T, O] = hall[B*T, H] @ W_dec[O, H]^T + b_dec
  gemm_bf16_kernel<<<dim3(O / 64, (Bsz * T) / 128), 256, 0, stream>>>(
      hall, (long)H, wdec_bf, (long)H, b_dec, out_o, (long)O, H);
}
